// ResidualAttentionBlock_39805756899968
// MI455X (gfx1250) — matrix operand order repaired and checked by reading, not compiled, not run on hardware
//
#include <hip/hip_runtime.h>
#include <hip/hip_bf16.h>
#include <math.h>

// Shapes (compile-time constants from the reference)
#define BB 4
#define TT 512
#define DD 256
#define HH 8
#define NQKV (3 * HH * DD)   // 6144
#define MROWS (BB * TT)      // 2048
#define SCALE 0.0625f        // 1/sqrt(256)

typedef __attribute__((ext_vector_type(16))) _Float16 v16h;
typedef __attribute__((ext_vector_type(8)))  float    v8f;
typedef __attribute__((ext_vector_type(4)))  unsigned int v4u;
typedef __attribute__((ext_vector_type(8)))  int      v8i;
typedef __attribute__((ext_vector_type(4)))  int      v4i;

#if __has_builtin(__builtin_amdgcn_tensor_load_to_lds)
#define HAVE_TDM 1
#else
#define HAVE_TDM 0
#endif

// ---------------------------------------------------------------------------
// WMMA fragment helpers, per CDNA5 ISA 7.12.2 (wave32 layouts)
// A (16x32 f16): lane = khalf*16 + m; VGPR v holds K = 2v + khalf*16 {+0,+1} (contiguous k, not ISA table order)
// B (32x16 f16): lane = khalf*16 + n; VGPR v holds K = 2v + khalf*16 {+0,+1} (contiguous k, not ISA table order)
// C/D (16x16 f32): lane = mhalf*16 + n; VGPR r holds M = r + mhalf*8
// ---------------------------------------------------------------------------

__device__ __forceinline__ v16h load_a_frag_f32(const float* __restrict__ src,
                                                int ld, int row0, int k0) {
  int lane = threadIdx.x & 31;
  int m = lane & 15;
  int khalf = lane >> 4;
  const float* p = src + (size_t)(row0 + m) * ld + k0;
  v16h a;
#pragma unroll
  for (int v = 0; v < 8; ++v) {
    int kb = (khalf << 4) + (v << 1);
    a[2 * v]     = (_Float16)p[kb];
    a[2 * v + 1] = (_Float16)p[kb + 1];
  }
  return a;
}

__device__ __forceinline__ v16h load_a_frag_f16(const _Float16* __restrict__ src,
                                                int ld, int row0, int k0) {
  int lane = threadIdx.x & 31;
  int m = lane & 15;
  int khalf = lane >> 4;
  const _Float16* p = src + (size_t)(row0 + m) * ld + k0;
  v16h a;
#pragma unroll
  for (int v = 0; v < 8; ++v) {
    int kb = (khalf << 4) + (v << 1);
    a[2 * v]     = p[kb];
    a[2 * v + 1] = p[kb + 1];
  }
  return a;
}

// B operand stored row-major K x N (f16): B[k][n] = src[k*ld + n]
__device__ __forceinline__ v16h load_b_frag_f16_kn(const _Float16* __restrict__ src,
                                                   int ld, int k0, int n0) {
  int lane = threadIdx.x & 31;
  int n = lane & 15;
  int koff = (lane >> 4) << 4;  // 0 or 16
  v16h b;
#pragma unroll
  for (int v = 0; v < 8; ++v) {
    const _Float16* p = src + (size_t)(k0 + koff + 2 * v) * ld + n0 + n;
    b[2 * v]     = p[0];
    b[2 * v + 1] = p[ld];
  }
  return b;
}

// B operand from an LDS panel stored row-major K x 16 (f16)
__device__ __forceinline__ v16h load_b_frag_lds(const _Float16* lds, int k0) {
  int lane = threadIdx.x & 31;
  int n = lane & 15;
  int koff = (lane >> 4) << 4;  // 0 or 16
  const _Float16* p = lds + (size_t)(k0 + koff) * 16 + n;
  v16h b;
#pragma unroll
  for (int v = 0; v < 8; ++v) {
    b[2 * v]     = p[(2 * v) * 16];
    b[2 * v + 1] = p[(2 * v + 1) * 16];
  }
  return b;
}

// B operand where B[k][n] = W[n][k], W f32 row-major N x K (for y = x @ W^T)
__device__ __forceinline__ v16h load_b_frag_f32_nk(const float* __restrict__ w,
                                                   int ld, int k0, int n0) {
  int lane = threadIdx.x & 31;
  int n = lane & 15;
  int koff = (lane >> 4) << 4;
  const float* p = w + (size_t)(n0 + n) * ld + k0 + koff;
  v16h b;
#pragma unroll
  for (int v = 0; v < 8; ++v) {
    b[2 * v]     = (_Float16)p[2 * v];
    b[2 * v + 1] = (_Float16)p[2 * v + 1];
  }
  return b;
}

#define WMMA_F16(a, b, c) \
  __builtin_amdgcn_wmma_f32_16x16x32_f16(false, (a), false, (b), (short)0, (c), false, false)

// ---------------------------------------------------------------------------
// TDM: issue a 2D tensor_load_to_lds for a (tile_h x tile_w) f16 tile out of a
// row-major tensor with row stride `stride` elements. D# layout per ISA §8.
// 6-arg builtin form: (uint32x4 g0, int32x8 g1, int32x4, int32x4, int32x8, cpol)
// ---------------------------------------------------------------------------
#if HAVE_TDM
__device__ __forceinline__ void tdm_load_tile_f16(const _Float16* gsrc,
                                                  unsigned lds_off,
                                                  unsigned tile_w, unsigned tile_h,
                                                  unsigned row_len, unsigned nrows,
                                                  unsigned stride) {
  unsigned long long ga = (unsigned long long)(uintptr_t)gsrc;
  v4u g0;
  g0[0] = 1u;                                             // count=1 (user D#)
  g0[1] = lds_off;                                        // lds_addr [63:32]
  g0[2] = (unsigned)ga;                                   // global_addr lo
  g0[3] = (unsigned)((ga >> 32) & 0x01FFFFFFu) | (2u << 30); // addr hi | type=2
  v8i g1;
  g1[0] = (int)(1u << 16);                 // workgroup_mask=0, data_size=1 (2B)
  g1[1] = (int)(row_len << 16);            // tensor_dim0[15:0] @ bits 63:48
  g1[2] = (int)((row_len >> 16) | (nrows << 16));   // dim0 hi | tensor_dim1 lo
  g1[3] = (int)((nrows >> 16) | (tile_w << 16));    // dim1 hi | tile_dim0
  g1[4] = (int)tile_h;                     // tile_dim1 (tile_dim2 = 0)
  g1[5] = (int)stride;                     // tensor_dim0_stride lo32
  g1[6] = 0;                               // stride hi | tensor_dim1_stride lo
  g1[7] = 0;
  v4i z4 = {0, 0, 0, 0};
  v8i z8 = {0, 0, 0, 0, 0, 0, 0, 0};
  __builtin_amdgcn_tensor_load_to_lds(g0, g1, z4, z4, z8, 0);
}
#endif

// ---------------------------------------------------------------------------
// K0: v = x @ wqv_w^T + wqv_b, routed into q (f32, [b,h,t,d]) / vf16 / vb16
// ---------------------------------------------------------------------------
__global__ void qkv_kernel(const float* __restrict__ x,
                           const float* __restrict__ wqv_w,
                           const float* __restrict__ wqv_b,
                           float* __restrict__ q,
                           _Float16* __restrict__ vf,
                           _Float16* __restrict__ vb) {
  int tile = blockIdx.x * 8 + (threadIdx.x >> 5);
  int tm = tile / (NQKV / 16);
  int tn = tile % (NQKV / 16);
  int row0 = tm * 16, col0 = tn * 16;

  v8f acc = {};
#pragma unroll
  for (int k0 = 0; k0 < DD; k0 += 32) {
    v16h af = load_a_frag_f32(x, DD, row0, k0);
    v16h bf = load_b_frag_f32_nk(wqv_w, DD, k0, col0);
    acc = WMMA_F16(af, bf, acc);
  }

  int lane = threadIdx.x & 31;
  int n = lane & 15;
  int moff = (lane >> 4) << 3;
  int col = col0 + n;
  int hh = col >> 8;
  int d  = col & 255;
  float bias = wqv_b[col];
#pragma unroll
  for (int r = 0; r < 8; ++r) {
    int row = row0 + moff + r;          // b*T + t
    int b = row >> 9, t = row & 511;
    float val = acc[r] + bias;
    size_t o = (((size_t)b * HH + (hh & 7)) * TT + t) * DD + d;
    if (hh < HH)            q[o]  = val;
    else if (hh < 2 * HH)   vf[o] = (_Float16)val;
    else                    vb[o] = (_Float16)val;
  }
}

// ---------------------------------------------------------------------------
// K1: a[b,h,i,j] = -scale * sum_d |q[b,h,i,d] - x[b,j,d]*wk[h,d]|  (VALU-bound)
// ---------------------------------------------------------------------------
__global__ void score_kernel(const float* __restrict__ x,
                             const float* __restrict__ wk,
                             const float* __restrict__ q,
                             float* __restrict__ a) {
  __shared__ __align__(16) float qs[16 * DD];
  __shared__ __align__(16) float ks[16 * DD];

  int idx = blockIdx.x;
  int bh = idx >> 10;
  int rest = idx & 1023;
  int ti = rest >> 5, tj = rest & 31;
  int b = bh >> 3, h = bh & 7;
  int tid = threadIdx.x;

  const float* qbase = q + ((size_t)bh * TT + ti * 16) * DD;
  for (int t = tid; t < 16 * DD; t += 256) qs[t] = qbase[t];
  for (int t = tid; t < 16 * DD; t += 256) {
    int r = t >> 8, c = t & 255;
    ks[t] = x[((size_t)b * TT + tj * 16 + r) * DD + c] * wk[h * DD + c];
  }
  __syncthreads();

  int i = tid >> 4, j = tid & 15;
  const float4* q4 = (const float4*)(qs + i * DD);
  const float4* k4 = (const float4*)(ks + j * DD);
  float s = 0.f;
#pragma unroll 4
  for (int dd = 0; dd < DD / 4; ++dd) {
    float4 qv = q4[dd], kv = k4[dd];
    s += fabsf(qv.x - kv.x) + fabsf(qv.y - kv.y) +
         fabsf(qv.z - kv.z) + fabsf(qv.w - kv.w);
  }
  a[((size_t)bh * TT + ti * 16 + i) * TT + tj * 16 + j] = -SCALE * s;
}

// ---------------------------------------------------------------------------
// K2: softmax over queries i (axis=1) -> afT[b,h,j,i] (f16, transposed store)
// ---------------------------------------------------------------------------
__global__ void colsoftmax_kernel(const float* __restrict__ a,
                                  _Float16* __restrict__ afT) {
  int c = blockIdx.x * 256 + threadIdx.x;   // 0 .. B*H*T-1
  int bh = c >> 9, j = c & 511;
  const float* base = a + (size_t)bh * TT * TT + j;
  float m = -INFINITY;
  for (int i = 0; i < TT; ++i) m = fmaxf(m, base[(size_t)i * TT]);
  float s = 0.f;
  for (int i = 0; i < TT; ++i) s += __expf(base[(size_t)i * TT] - m);
  float inv = 1.0f / s;
  _Float16* o = afT + ((size_t)bh * TT + j) * TT;
  for (int i = 0; i < TT; ++i)
    o[i] = (_Float16)(__expf(base[(size_t)i * TT] - m) * inv);
}

// ---------------------------------------------------------------------------
// K3: softmax over keys j (axis=2) -> ab[b,h,i,j] (f16)
// ---------------------------------------------------------------------------
__global__ void rowsoftmax_kernel(const float* __restrict__ a,
                                  _Float16* __restrict__ ab) {
  int r = blockIdx.x * 256 + threadIdx.x;
  int bh = r >> 9, i = r & 511;
  const float* base = a + ((size_t)bh * TT + i) * TT;
  float m = -INFINITY;
  for (int j = 0; j < TT; ++j) m = fmaxf(m, base[j]);
  float s = 0.f;
  for (int j = 0; j < TT; ++j) s += __expf(base[j] - m);
  float inv = 1.0f / s;
  _Float16* o = ab + ((size_t)bh * TT + i) * TT;
  for (int j = 0; j < TT; ++j) o[j] = (_Float16)(__expf(base[j] - m) * inv);
}

// ---------------------------------------------------------------------------
// K4: bo = sum_h (afT @ vf + ab @ vb); g = bo*sigmoid(1.702*bo)
// Block = 8 waves sharing one (b, n-tile) column over 8 consecutive m-tiles.
// vf/vb 512x16 panels are staged into LDS by the Tensor Data Mover (wave 0
// issues tensor_load_to_lds, waits TENSORcnt, then workgroup barrier); B
// fragments come from LDS, A fragments from global (unique per wave).
// ---------------------------------------------------------------------------
__global__ void attn_kernel(const _Float16* __restrict__ afT,
                            const _Float16* __restrict__ ab,
                            const _Float16* __restrict__ vf,
                            const _Float16* __restrict__ vb,
                            _Float16* __restrict__ g) {
  __shared__ __align__(16) _Float16 vfs[TT * 16];   // 16 KB
  __shared__ __align__(16) _Float16 vbs[TT * 16];   // 16 KB

  int blk = blockIdx.x;           // 0..255
  int b   = blk >> 6;             // batch
  int tn  = (blk >> 2) & 15;      // n-tile (w)
  int tmg = blk & 3;              // m-tile group
  int tm  = tmg * 8 + (threadIdx.x >> 5);
  int row0 = tm * 16, col0 = tn * 16;

  v8f acc = {};
  for (int h = 0; h < HH; ++h) {
    const _Float16* afb = afT + (size_t)(b * HH + h) * TT * TT;
    const _Float16* abb = ab  + (size_t)(b * HH + h) * TT * TT;
    const _Float16* vfb = vf  + (size_t)(b * HH + h) * TT * DD + col0;
    const _Float16* vbb = vb  + (size_t)(b * HH + h) * TT * DD + col0;

    __syncthreads();   // previous-h panel reads complete before overwrite
#if HAVE_TDM
    if (threadIdx.x < 32) {        // one wave drives the TDM
      tdm_load_tile_f16(vfb, (unsigned)(uintptr_t)vfs, 16, TT, DD, TT, DD);
      tdm_load_tile_f16(vbb, (unsigned)(uintptr_t)vbs, 16, TT, DD, TT, DD);
      __builtin_amdgcn_s_wait_tensorcnt(0);
    }
#else
    for (int t = threadIdx.x; t < TT * 16; t += 256) {
      int kk = t >> 4, n = t & 15;
      vfs[t] = vfb[(size_t)kk * DD + n];
      vbs[t] = vbb[(size_t)kk * DD + n];
    }
#endif
    __syncthreads();

#pragma unroll 2
    for (int k0 = 0; k0 < TT; k0 += 32) {
      if (k0 + 32 < TT)  // warm GL2 for the next K-slab (global_prefetch_b8)
        __builtin_prefetch(afb + (size_t)row0 * TT + k0 + 32, 0, 3);
      v16h a1 = load_a_frag_f16(afb, TT, row0, k0);
      v16h b1 = load_b_frag_lds(vfs, k0);
      acc = WMMA_F16(a1, b1, acc);
      v16h a2 = load_a_frag_f16(abb, TT, row0, k0);
      v16h b2 = load_b_frag_lds(vbs, k0);
      acc = WMMA_F16(a2, b2, acc);
    }
  }

  int lane = threadIdx.x & 31;
  int n = lane & 15;
  int moff = (lane >> 4) << 3;
#pragma unroll
  for (int r = 0; r < 8; ++r) {
    int t = row0 + moff + r;
    float v = acc[r];
    float y = v / (1.0f + __expf(-1.702f * v));   // v * sigmoid(1.702 v)
    g[((size_t)b * TT + t) * DD + col0 + n] = (_Float16)y;
  }
}

// ---------------------------------------------------------------------------
// K5: out = x + g @ fanin_w^T + fanin_b   (M=2048, N=256, K=256)
// ---------------------------------------------------------------------------
__global__ void fanin_kernel(const _Float16* __restrict__ g,
                             const float* __restrict__ fanin_w,
                             const float* __restrict__ fanin_b,
                             const float* __restrict__ x,
                             float* __restrict__ out) {
  int tile = blockIdx.x * 8 + (threadIdx.x >> 5);   // 0..2047
  int tm = tile >> 4, tn = tile & 15;
  int row0 = tm * 16, col0 = tn * 16;

  v8f acc = {};
#pragma unroll
  for (int k0 = 0; k0 < DD; k0 += 32) {
    v16h af = load_a_frag_f16(g, DD, row0, k0);
    v16h bf = load_b_frag_f32_nk(fanin_w, DD, k0, col0);
    acc = WMMA_F16(af, bf, acc);
  }

  int lane = threadIdx.x & 31;
  int n = lane & 15;
  int moff = (lane >> 4) << 3;
#pragma unroll
  for (int r = 0; r < 8; ++r) {
    size_t o = (size_t)(row0 + moff + r) * DD + col0 + n;
    out[o] = x[o] + acc[r] + fanin_b[col0 + n];
  }
}

// ---------------------------------------------------------------------------
extern "C" void kernel_launch(void* const* d_in, const int* in_sizes, int n_in,
                              void* d_out, int out_size, void* d_ws, size_t ws_size,
                              hipStream_t stream) {
  const float* x       = (const float*)d_in[0];
  const float* wk      = (const float*)d_in[1];
  const float* wqv_w   = (const float*)d_in[2];
  const float* wqv_b   = (const float*)d_in[3];
  const float* fanin_w = (const float*)d_in[4];
  const float* fanin_b = (const float*)d_in[5];
  // d_in[6] = use_dp (ignored; reference ignores it too)
  float* out = (float*)d_out;

  char* ws = (char*)d_ws;
  float*    q    = (float*)ws;     ws += (size_t)BB * HH * TT * DD * 4;  // 16 MB
  _Float16* vf   = (_Float16*)ws;  ws += (size_t)BB * HH * TT * DD * 2;  //  8 MB
  _Float16* vb   = (_Float16*)ws;  ws += (size_t)BB * HH * TT * DD * 2;  //  8 MB
  float*    a    = (float*)ws;     ws += (size_t)BB * HH * TT * TT * 4;  // 32 MB
  _Float16* afT  = (_Float16*)ws;  ws += (size_t)BB * HH * TT * TT * 2;  // 16 MB
  _Float16* ab   = (_Float16*)ws;  ws += (size_t)BB * HH * TT * TT * 2;  // 16 MB
  _Float16* g    = (_Float16*)ws;  ws += (size_t)MROWS * DD * 2;         //  1 MB

  qkv_kernel<<<(MROWS / 16) * (NQKV / 16) / 8, 256, 0, stream>>>(
      x, wqv_w, wqv_b, q, vf, vb);

  score_kernel<<<BB * HH * (TT / 16) * (TT / 16), 256, 0, stream>>>(x, wk, q, a);

  colsoftmax_kernel<<<BB * HH * TT / 256, 256, 0, stream>>>(a, afT);
  rowsoftmax_kernel<<<BB * HH * TT / 256, 256, 0, stream>>>(a, ab);

  attn_kernel<<<BB * (TT / 16) * (DD / 16) / 8, 256, 0, stream>>>(
      afT, ab, vf, vb, g);

  fanin_kernel<<<(MROWS / 16) * (DD / 16) / 8, 256, 0, stream>>>(
      g, fanin_w, fanin_b, x, out);
}
